// AttentionTalkingHead_5377299055029
// MI455X (gfx1250) — matrix operand order repaired and checked by reading, not compiled, not run on hardware
//
#include <hip/hip_runtime.h>
#include <hip/hip_bf16.h>

// ---------------------------------------------------------------------------
// Talking-heads attention, fused two-pass flash style, bf16 WMMA on gfx1250.
//   B=4, S=2048, D=512, H=8, hd=64, SCALE = 1/8
// ---------------------------------------------------------------------------

#define BATCH    4
#define SEQ      2048
#define DIM      512
#define HEADS    8
#define HEAD_DIM 64
#define QK_SCALE 0.125f

typedef __attribute__((ext_vector_type(16))) __bf16 bf16x16;
typedef __attribute__((ext_vector_type(8)))  __bf16 bf16x8;
typedef __attribute__((ext_vector_type(8)))  float  f32x8;

static __device__ __forceinline__ bf16x16 make_frag(bf16x8 lo, bf16x8 hi) {
    union { bf16x16 v; bf16x8 h[2]; } u;
    u.h[0] = lo; u.h[1] = hi;
    return u.v;
}

static __device__ __forceinline__ f32x8 wmma_bf16(bf16x16 a, bf16x16 b, f32x8 c) {
    // D = A(16x32 bf16) * B(32x16 bf16) + C(16x16 f32)
    return __builtin_amdgcn_wmma_f32_16x16x32_bf16(
        /*neg_a=*/false, a, /*neg_b=*/false, b,
        /*c_mod=*/(short)0, c, /*reuse_a=*/false, /*reuse_b=*/false);
}

// ---------------------------------------------------------------------------
// Kernel 1: f32 -> bf16 cast (grid-stride)
// ---------------------------------------------------------------------------
__global__ void cvt_f32_bf16(const float* __restrict__ in,
                             __bf16* __restrict__ out, int n) {
    int i = blockIdx.x * blockDim.x + threadIdx.x;
    int stride = gridDim.x * blockDim.x;
    for (; i < n; i += stride) out[i] = (__bf16)in[i];
}

// ---------------------------------------------------------------------------
// Kernel 2: QKV projection GEMM.
//   C[M=8192, N=1536] = Xbf[8192x512] @ Wqkv[1536x512]^T   (K contiguous both)
//   Epilogue scatters to Q [B,H,S,hd] (pre-scaled), K [B,H,S,hd], VT [B,H,hd,S].
//   4 waves / block, each wave owns one 16x16 tile of a 16x64 block tile.
// ---------------------------------------------------------------------------
__global__ __launch_bounds__(128)
void qkv_gemm(const __bf16* __restrict__ X, const __bf16* __restrict__ W,
              __bf16* __restrict__ Q, __bf16* __restrict__ K,
              __bf16* __restrict__ VT) {
    const int lane = threadIdx.x & 31;
    const int wave = threadIdx.x >> 5;
    const int half = lane >> 4;
    const int l16  = lane & 15;
    const int m0 = blockIdx.x * 16;
    const int n0 = blockIdx.y * 64 + wave * 16;

    const __bf16* arow = X + (size_t)(m0 + l16) * DIM;  // A: lane = M row
    const __bf16* brow = W + (size_t)(n0 + l16) * DIM;  // B: lane = N col

    f32x8 acc = {};
    for (int kk = 0; kk < DIM; kk += 32) {
        bf16x8  alo = *(const bf16x8*)(arow + kk + 16 * half);
        bf16x8  ahi = *(const bf16x8*)(arow + kk + 8 + 16 * half);
        bf16x16 bfr = *(const bf16x16*)(brow + kk + 16 * half);
        acc = wmma_bf16(make_frag(alo, ahi), bfr, acc);
    }

    const int sec = n0 / DIM;            // 0=Q 1=K 2=V
    const int nin = n0 % DIM;
    const int h   = nin / HEAD_DIM;
    const int d   = (nin % HEAD_DIM) + l16;

    for (int r = 0; r < 8; ++r) {
        const int m = m0 + 8 * half + r;
        const int b = m >> 11, s = m & (SEQ - 1);
        const size_t bh = (size_t)(b * HEADS + h);
        const float v = acc[r];
        if (sec == 0)      Q[(bh * SEQ + s) * HEAD_DIM + d] = (__bf16)(v * QK_SCALE);
        else if (sec == 1) K[(bh * SEQ + s) * HEAD_DIM + d] = (__bf16)v;
        else               VT[(bh * HEAD_DIM + d) * SEQ + s] = (__bf16)v;
    }
}

// ---------------------------------------------------------------------------
// Kernel 3: fused talking-heads attention, two passes over keys.
//   Block = 256 threads = 8 waves; wave h owns head h of one (batch, 16-query)
//   tile. Pass 1: mixed-score row max/sum (online). Pass 2: recompute scores,
//   normalize, post-mix, P@V via WMMA. Cross-head exchange through LDS.
// ---------------------------------------------------------------------------
__global__ __launch_bounds__(256)
void attn_talking_heads(const __bf16* __restrict__ Q, const __bf16* __restrict__ K,
                        const __bf16* __restrict__ VT,
                        const float* __restrict__ w_l, const float* __restrict__ b_l,
                        const float* __restrict__ w_w, const float* __restrict__ b_w,
                        __bf16* __restrict__ AO) {
    __shared__ float  lds_raw[HEADS][16 * 64];   // raw scores, reused for attn2(bf16)
    __shared__ __bf16 lds_p[HEADS][16 * 64];     // normalized probabilities
    __shared__ float  lds_wl[64], lds_ww[64], lds_bl[8], lds_bw[8];

    const int tid  = threadIdx.x;
    const int lane = tid & 31;
    const int wave = tid >> 5;                   // == head index
    const int half = lane >> 4;
    const int l16  = lane & 15;
    const int b    = blockIdx.x >> 7;            // 128 q-tiles per batch
    const int q0   = (blockIdx.x & 127) * 16;

    if (tid < 64) { lds_wl[tid] = w_l[tid]; lds_ww[tid] = w_w[tid]; }
    if (tid < 8)  { lds_bl[tid] = b_l[tid]; lds_bw[tid] = b_w[tid]; }
    __syncthreads();

    const int h = wave;
    const size_t bh = (size_t)(b * HEADS + h);

    // Q A-fragments (hd = K dim of QK^T), resident for whole kernel.
    const __bf16* qrow = Q + (bh * SEQ + q0 + l16) * HEAD_DIM;
    const bf16x16 aq0 = make_frag(*(const bf16x8*)(qrow + 16 * half),
                                  *(const bf16x8*)(qrow + 8 + 16 * half));
    const bf16x16 aq1 = make_frag(*(const bf16x8*)(qrow + 32 + 16 * half),
                                  *(const bf16x8*)(qrow + 40 + 16 * half));

    float wlr[8], wwr[8];
    for (int i = 0; i < 8; ++i) { wlr[i] = lds_wl[h * 8 + i]; wwr[i] = lds_ww[h * 8 + i]; }
    const float blh = lds_bl[h];
    const float bwh = lds_bw[h];

    const __bf16* Kbase  = K + bh * SEQ * HEAD_DIM;
    const __bf16* VTbase = VT + bh * (size_t)HEAD_DIM * SEQ;

    float m[8], l[8];
    for (int r = 0; r < 8; ++r) { m[r] = -1e30f; l[r] = 0.0f; }

    // ---------------- pass 1: mixed-score softmax statistics ----------------
    for (int kb = 0; kb < SEQ; kb += 64) {
        f32x8 acc[4];
        for (int j = 0; j < 4; ++j) {
            const __bf16* krow = Kbase + (size_t)(kb + 16 * j + l16) * HEAD_DIM;
            bf16x16 bk0 = *(const bf16x16*)(krow + 16 * half);
            bf16x16 bk1 = *(const bf16x16*)(krow + 32 + 16 * half);
            f32x8 c = {};
            c = wmma_bf16(aq0, bk0, c);
            c = wmma_bf16(aq1, bk1, c);
            acc[j] = c;
        }
        for (int j = 0; j < 4; ++j)
            for (int r = 0; r < 8; ++r)
                lds_raw[h][(8 * half + r) * 64 + 16 * j + l16] = acc[j][r];
        __syncthreads();

        float smix[4][8];
        for (int j = 0; j < 4; ++j)
            for (int r = 0; r < 8; ++r) {
                const int idx = (8 * half + r) * 64 + 16 * j + l16;
                float s = blh;
                for (int h2 = 0; h2 < 8; ++h2) s += wlr[h2] * lds_raw[h2][idx];
                smix[j][r] = s;
            }
        for (int r = 0; r < 8; ++r) {
            float mx = smix[0][r];
            for (int j = 1; j < 4; ++j) mx = fmaxf(mx, smix[j][r]);
            for (int off = 1; off < 16; off <<= 1)   // reduce over 16-lane half
                mx = fmaxf(mx, __shfl_xor(mx, off, 32));
            const float mn = fmaxf(m[r], mx);
            float ssum = 0.0f;
            for (int j = 0; j < 4; ++j) ssum += __expf(smix[j][r] - mn);
            for (int off = 1; off < 16; off <<= 1)
                ssum += __shfl_xor(ssum, off, 32);
            l[r] = l[r] * __expf(m[r] - mn) + ssum;
            m[r] = mn;
        }
        __syncthreads();
    }
    float linv[8];
    for (int r = 0; r < 8; ++r) linv[r] = 1.0f / l[r];

    // ---------------- pass 2: normalize, post-mix, P @ V --------------------
    __bf16* lds_a = (__bf16*)&lds_raw[h][0];   // attn2 tile for this head
    f32x8 o[4] = {};

    for (int kb = 0; kb < SEQ; kb += 64) {
        f32x8 acc[4];
        for (int j = 0; j < 4; ++j) {
            const __bf16* krow = Kbase + (size_t)(kb + 16 * j + l16) * HEAD_DIM;
            bf16x16 bk0 = *(const bf16x16*)(krow + 16 * half);
            bf16x16 bk1 = *(const bf16x16*)(krow + 32 + 16 * half);
            f32x8 c = {};
            c = wmma_bf16(aq0, bk0, c);
            c = wmma_bf16(aq1, bk1, c);
            acc[j] = c;
        }
        for (int j = 0; j < 4; ++j)
            for (int r = 0; r < 8; ++r)
                lds_raw[h][(8 * half + r) * 64 + 16 * j + l16] = acc[j][r];
        __syncthreads();

        // P[h] = softmax_h(mixed scores), using pass-1 stats held in registers
        for (int j = 0; j < 4; ++j)
            for (int r = 0; r < 8; ++r) {
                const int idx = (8 * half + r) * 64 + 16 * j + l16;
                float s = blh;
                for (int h2 = 0; h2 < 8; ++h2) s += wlr[h2] * lds_raw[h2][idx];
                lds_p[h][idx] = (__bf16)(__expf(s - m[r]) * linv[r]);
            }
        __syncthreads();

        // attn2[h] = b_w[h] + sum_g w_w[h,g] * P[g]; pack bf16 into lds_raw[h]
        for (int j = 0; j < 4; ++j)
            for (int r = 0; r < 8; ++r) {
                const int idx = (8 * half + r) * 64 + 16 * j + l16;
                float a2 = bwh;
                for (int h2 = 0; h2 < 8; ++h2) a2 += wwr[h2] * (float)lds_p[h2][idx];
                lds_a[idx] = (__bf16)a2;
            }

        // A-fragments of attn2 (same-wave LDS RAW -> in-order, no barrier)
        bf16x16 ap[2];
        for (int c = 0; c < 2; ++c) {
            const __bf16* ar = lds_a + l16 * 64 + 32 * c;
            ap[c] = make_frag(*(const bf16x8*)(ar + 16 * half),
                              *(const bf16x8*)(ar + 8 + 16 * half));
        }
        // O += attn2(16x64) @ V(64x64); V^T layout makes B-frags contiguous
        for (int j = 0; j < 4; ++j)
            for (int c = 0; c < 2; ++c) {
                const __bf16* vr = VTbase + (size_t)(16 * j + l16) * SEQ
                                 + kb + 32 * c + 16 * half;
                bf16x16 bv = *(const bf16x16*)vr;
                o[j] = wmma_bf16(ap[c], bv, o[j]);
            }
        __syncthreads();
    }

    // head-concat output tile -> AO [B, S, D] bf16
    for (int j = 0; j < 4; ++j)
        for (int r = 0; r < 8; ++r) {
            const int srow = q0 + 8 * half + r;
            AO[((size_t)(b * SEQ + srow)) * DIM + h * HEAD_DIM + 16 * j + l16] =
                (__bf16)o[j][r];
        }
}

// ---------------------------------------------------------------------------
// Kernel 4: output projection GEMM + bias -> f32 d_out.
//   out[M=8192, N=512] = AO[8192x512] @ Wproj[512x512]^T + b_proj
// ---------------------------------------------------------------------------
__global__ __launch_bounds__(128)
void proj_gemm(const __bf16* __restrict__ A, const __bf16* __restrict__ W,
               const float* __restrict__ bias, float* __restrict__ out) {
    const int lane = threadIdx.x & 31;
    const int wave = threadIdx.x >> 5;
    const int half = lane >> 4;
    const int l16  = lane & 15;
    const int m0 = blockIdx.x * 16;
    const int n0 = blockIdx.y * 64 + wave * 16;

    const __bf16* arow = A + (size_t)(m0 + l16) * DIM;
    const __bf16* brow = W + (size_t)(n0 + l16) * DIM;

    f32x8 acc = {};
    for (int kk = 0; kk < DIM; kk += 32) {
        bf16x8  alo = *(const bf16x8*)(arow + kk + 16 * half);
        bf16x8  ahi = *(const bf16x8*)(arow + kk + 8 + 16 * half);
        bf16x16 bfr = *(const bf16x16*)(brow + kk + 16 * half);
        acc = wmma_bf16(make_frag(alo, ahi), bfr, acc);
    }

    const int n = n0 + l16;
    const float bn = bias[n];
    for (int r = 0; r < 8; ++r)
        out[(size_t)(m0 + 8 * half + r) * DIM + n] = acc[r] + bn;
}

// ---------------------------------------------------------------------------
extern "C" void kernel_launch(void* const* d_in, const int* in_sizes, int n_in,
                              void* d_out, int out_size, void* d_ws, size_t ws_size,
                              hipStream_t stream) {
    (void)in_sizes; (void)n_in; (void)out_size; (void)ws_size;
    const float* x      = (const float*)d_in[0];
    const float* w_qkv  = (const float*)d_in[1];
    const float* w_proj = (const float*)d_in[2];
    const float* b_proj = (const float*)d_in[3];
    const float* w_l    = (const float*)d_in[4];
    const float* b_l    = (const float*)d_in[5];
    const float* w_w    = (const float*)d_in[6];
    const float* b_w    = (const float*)d_in[7];

    const size_t M = (size_t)BATCH * SEQ;              // 8192
    char* ws = (char*)d_ws;
    size_t off = 0;
    auto carve = [&](size_t bytes) -> char* {
        char* p = ws + off;
        off += (bytes + 255) & ~(size_t)255;
        return p;
    };
    __bf16* xb     = (__bf16*)carve(M * DIM * 2);                  // 8 MB
    __bf16* wqkvb  = (__bf16*)carve((size_t)3 * DIM * DIM * 2);    // 1.5 MB
    __bf16* wprojb = (__bf16*)carve((size_t)DIM * DIM * 2);        // 0.5 MB
    __bf16* Qb     = (__bf16*)carve(M * DIM * 2);                  // 8 MB
    __bf16* Kb     = (__bf16*)carve(M * DIM * 2);                  // 8 MB
    __bf16* VTb    = (__bf16*)carve(M * DIM * 2);                  // 8 MB
    __bf16* AOb    = (__bf16*)carve(M * DIM * 2);                  // 8 MB

    cvt_f32_bf16<<<1024, 256, 0, stream>>>(x, xb, (int)(M * DIM));
    cvt_f32_bf16<<<512, 256, 0, stream>>>(w_qkv, wqkvb, 3 * DIM * DIM);
    cvt_f32_bf16<<<256, 256, 0, stream>>>(w_proj, wprojb, DIM * DIM);

    qkv_gemm<<<dim3((unsigned)(M / 16), 3 * DIM / 64), 128, 0, stream>>>(
        xb, wqkvb, Qb, Kb, VTb);

    attn_talking_heads<<<dim3((unsigned)(BATCH * SEQ / 16)), 256, 0, stream>>>(
        Qb, Kb, VTb, w_l, b_l, w_w, b_w, AOb);

    proj_gemm<<<dim3((unsigned)(M / 16), DIM / 64), 128, 0, stream>>>(
        AOb, wprojb, b_proj, (float*)d_out);
}
